// BlockPointNet_29532195127624
// MI455X (gfx1250) — matrix operand order repaired and checked by reading, not compiled, not run on hardware
//
#include <hip/hip_runtime.h>
#include <hip/hip_bf16.h>

// ---------------------------------------------------------------------------
// PointNet++-style two-branch network for MI455X (gfx1250, wave32, WMMA).
// MLP GEMMs: v_wmma_f32_16x16x32_f16, f32 accumulate, async-LDS-staged weights,
// b128 A-fragment loads, raw split barriers (no storecnt drain per tile).
// ---------------------------------------------------------------------------

typedef __attribute__((ext_vector_type(16))) _Float16 v16h;
typedef __attribute__((ext_vector_type(8)))  float    v8f;
typedef __attribute__((ext_vector_type(4)))  unsigned v4u;

#define BN_RS 0.9999950000374997f   /* 1/sqrt(1+1e-5) */
#define F_INF 3.402823466e38f
#define NEG_INF (-3.0e38f)

// LDS-only workgroup barrier: async/ds ordering is already guaranteed by the
// explicit s_wait_asynccnt and the dscnt waits the compiler inserts before
// WMMA consumes LDS data, so skip __syncthreads()'s storecnt drain.
#define WG_BARRIER() \
    asm volatile("s_barrier_signal -1\n\ts_barrier_wait -1" ::: "memory")

union AFrag { v16h h; v4u q[2]; unsigned u[8]; };

// ------------------------------ FPS ---------------------------------------
// One block (1024 threads) per batch element; m-1 sequential argmax passes.
template<int N>
__global__ __launch_bounds__(1024)
void fps_kernel(const float* __restrict__ pos, int m, int* __restrict__ idx)
{
    __shared__ float dist[N];
    __shared__ float rv[1024];
    __shared__ int   ri[1024];
    __shared__ int   snxt;
    const int b = blockIdx.x;
    const int t = threadIdx.x;
    const float px0 = pos[(size_t)b * N * 3 + 0];
    const float py0 = pos[(size_t)b * N * 3 + 1];
    const float pz0 = pos[(size_t)b * N * 3 + 2];
    for (int i = t; i < N; i += 1024) {
        float dx = pos[((size_t)b * N + i) * 3 + 0] - px0;
        float dy = pos[((size_t)b * N + i) * 3 + 1] - py0;
        float dz = pos[((size_t)b * N + i) * 3 + 2] - pz0;
        dist[i] = dx * dx + dy * dy + dz * dz;
    }
    if (t == 0) idx[(size_t)b * m] = 0;
    for (int s = 1; s < m; ++s) {
        __syncthreads();
        float bv = -1.0f; int bi = 0;
        for (int i = t; i < N; i += 1024) {
            float v = dist[i];
            if (v > bv) { bv = v; bi = i; }
        }
        rv[t] = bv; ri[t] = bi;
        __syncthreads();
        for (int sh = 512; sh > 0; sh >>= 1) {
            if (t < sh) {
                if (rv[t + sh] > rv[t] ||
                    (rv[t + sh] == rv[t] && ri[t + sh] < ri[t])) {
                    rv[t] = rv[t + sh]; ri[t] = ri[t + sh];
                }
            }
            __syncthreads();
        }
        if (t == 0) { snxt = ri[0]; idx[(size_t)b * m + s] = ri[0]; }
        __syncthreads();
        const int nx = snxt;
        const float qx = pos[((size_t)b * N + nx) * 3 + 0];
        const float qy = pos[((size_t)b * N + nx) * 3 + 1];
        const float qz = pos[((size_t)b * N + nx) * 3 + 2];
        for (int i = t; i < N; i += 1024) {
            float dx = pos[((size_t)b * N + i) * 3 + 0] - qx;
            float dy = pos[((size_t)b * N + i) * 3 + 1] - qy;
            float dz = pos[((size_t)b * N + i) * 3 + 2] - qz;
            float nd = dx * dx + dy * dy + dz * dz;
            if (nd < dist[i]) dist[i] = nd;
        }
    }
}

__global__ void gather_centers_kernel(const float* __restrict__ pos,
                                      const int* __restrict__ idx,
                                      float* __restrict__ ctr, int N, int M)
{
    int i = blockIdx.x * 256 + threadIdx.x;
    if (i >= 8 * M * 3) return;
    int c = i % 3;
    int bm = i / 3;
    int b = bm / M;
    ctr[i] = pos[((size_t)b * N + idx[bm]) * 3 + c];
}

// ------------------------ Radius ball-query (k nearest) --------------------
template<int N>
__global__ __launch_bounds__(256)
void group_kernel(const float* __restrict__ pos, const float* __restrict__ ctr,
                  int M, float r2, int* __restrict__ nbr, int* __restrict__ cnt)
{
    __shared__ float d2[N];
    __shared__ float rv[256];
    __shared__ int   ri[256];
    __shared__ int   sdone;
    __shared__ int   scnt;
    const int bm = blockIdx.x;
    const int b = bm / M;
    const int t = threadIdx.x;
    const float cx = ctr[(size_t)bm * 3 + 0];
    const float cy = ctr[(size_t)bm * 3 + 1];
    const float cz = ctr[(size_t)bm * 3 + 2];
    for (int i = t; i < N; i += 256) {
        float dx = pos[((size_t)b * N + i) * 3 + 0] - cx;
        float dy = pos[((size_t)b * N + i) * 3 + 1] - cy;
        float dz = pos[((size_t)b * N + i) * 3 + 2] - cz;
        d2[i] = dx * dx + dy * dy + dz * dz;
    }
    if (t == 0) { sdone = 0; scnt = 0; }
    for (int j = 0; j < 64; ++j) {
        __syncthreads();
        float bv = F_INF; int bi = -1;
        if (!sdone) {
            for (int i = t; i < N; i += 256) {
                float v = d2[i];
                if (v < bv) { bv = v; bi = i; }
            }
        }
        rv[t] = bv; ri[t] = bi;
        __syncthreads();
        for (int sh = 128; sh > 0; sh >>= 1) {
            if (t < sh) {
                if (rv[t + sh] < rv[t] ||
                    (rv[t + sh] == rv[t] && ri[t + sh] >= 0 && ri[t + sh] < ri[t])) {
                    rv[t] = rv[t + sh]; ri[t] = ri[t + sh];
                }
            }
            __syncthreads();
        }
        if (t == 0) {
            int sel = -1;
            if (!sdone && ri[0] >= 0 && rv[0] <= r2) sel = ri[0];
            else sdone = 1;
            if (sel >= 0) {
                nbr[(size_t)bm * 64 + j] = sel;
                d2[sel] = F_INF;
                scnt++;
            } else {
                nbr[(size_t)bm * 64 + j] = 0;
            }
        }
    }
    __syncthreads();
    if (t == 0) cnt[bm] = scnt;
}

// ------------------------- Feature builders (f16 rows) ---------------------
__global__ void feat_sa1_kernel(const float* __restrict__ pos,
                                const float* __restrict__ ctr,
                                const int* __restrict__ nbr,
                                const int* __restrict__ cnt,
                                _Float16* __restrict__ F) // [8*1024*64, 32]
{
    size_t i = (size_t)blockIdx.x * 256 + threadIdx.x;
    if (i >= (size_t)8 * 1024 * 64 * 32) return;
    int c = (int)(i & 31);
    size_t row = i >> 5;
    int j = (int)(row & 63);
    size_t bm = row >> 6;
    float v = 0.0f;
    if (c < 3 && j < cnt[bm]) {
        int b = (int)(bm / 1024);
        int s = nbr[bm * 64 + j];
        v = pos[((size_t)b * 2048 + s) * 3 + c] - ctr[bm * 3 + c];
    }
    F[i] = (_Float16)v;
}

__global__ void feat_sa2_kernel(const _Float16* __restrict__ x1, // [8,1024,128]
                                const float* __restrict__ c1,    // [8,1024,3]
                                const float* __restrict__ c2,    // [8,256,3]
                                const int* __restrict__ nbr,
                                const int* __restrict__ cnt,
                                _Float16* __restrict__ F) // [8*256*64, 160]
{
    size_t i = (size_t)blockIdx.x * 256 + threadIdx.x;
    if (i >= (size_t)8 * 256 * 64 * 160) return;
    int c = (int)(i % 160);
    size_t row = i / 160;
    int j = (int)(row & 63);
    size_t bm = row >> 6;
    float v = 0.0f;
    if (j < cnt[bm]) {
        int b = (int)(bm / 256);
        int s = nbr[bm * 64 + j];
        if (c < 128) {
            v = (float)x1[((size_t)b * 1024 + s) * 128 + c];
        } else if (c < 131) {
            int d = c - 128;
            v = c1[((size_t)b * 1024 + s) * 3 + d] - c2[bm * 3 + d];
        }
    }
    F[i] = (_Float16)v;
}

__global__ void feat_sa3_kernel(const _Float16* __restrict__ x2, // [8,256,256]
                                const float* __restrict__ c2,    // [8,256,3]
                                _Float16* __restrict__ F)        // [2048, 288]
{
    size_t i = (size_t)blockIdx.x * 256 + threadIdx.x;
    if (i >= (size_t)2048 * 288) return;
    int c = (int)(i % 288);
    size_t row = i / 288;
    float v = 0.0f;
    if (c < 256) v = (float)x2[row * 256 + c];
    else if (c < 259) v = c2[row * 3 + (c - 256)];
    F[i] = (_Float16)v;
}

// --------------------------- WMMA MLP GEMM ---------------------------------
// Y[R,COUT] = BN(ReLU(X[R,CINP] @ W + b)).
// B-fragments staged global->LDS with async copies (ASYNCcnt, double-buffered,
// chunk nt+1 overlapped with compute of chunk nt). A-fragments: 2x b128 loads
// per chunk. Output tile transposed through per-wave LDS -> b128 stores.
template<int CINP, int COUT>
__global__ __launch_bounds__(256)
void mlp_gemm_kernel(const _Float16* __restrict__ X,
                     const _Float16* __restrict__ Wpk,
                     const float* __restrict__ bias,
                     const float* __restrict__ gamma,
                     const float* __restrict__ beta,
                     _Float16* __restrict__ Y)
{
    constexpr int KCH   = CINP / 32;
    constexpr int NT    = COUT / 16;
    constexpr int CHUNK = KCH * 1024;        // bytes of B per 16-col tile
    constexpr int SLICE = CHUNK / 8;         // bytes per wave per chunk
    constexpr int CPW   = (KCH + 3) / 4;     // async b128 instrs per wave/chunk

    __shared__ __align__(16) _Float16 bsh[2][KCH * 512];
    __shared__ __align__(16) _Float16 tsh[8][256];

    const int wave = threadIdx.x >> 5;
    const int lane = threadIdx.x & 31;
    const size_t row0 = ((size_t)blockIdx.x * 8 + wave) * 16;
    const size_t mrow = row0 + (lane & 15);
    const int kbase = (lane < 16) ? 0 : 8;

    // A fragments: documented 16-bit A 16x32 layout. Each lane's 16 halves are
    // two contiguous 16B runs (K=kbase..+7 and K=kbase+16..+23) -> 2x b128.
    const char* xrow =
        (const char*)X + ((size_t)mrow * CINP + kbase) * 2;
    AFrag a[KCH];
#pragma unroll
    for (int kc = 0; kc < KCH; ++kc) {
        a[kc].q[0] = *(const v4u*)(xrow + (size_t)kc * 64);
        a[kc].q[1] = *(const v4u*)(xrow + (size_t)kc * 64 + 32);
    }

    // async global->LDS copy of B chunk `nt` into buffer `bufi`
    auto issue = [&](int nt, int bufi) {
        const char* gsrc = (const char*)Wpk + (size_t)nt * CHUNK + wave * SLICE;
        const unsigned ldst =
            (unsigned)(size_t)&bsh[bufi][0] + (unsigned)(wave * SLICE);
#pragma unroll
        for (int r = 0; r < CPW; ++r) {
            int ofs = lane * 16 + r * 512;
            if (ofs < SLICE) {
                unsigned long long ga = (unsigned long long)(gsrc + ofs);
                unsigned la = ldst + (unsigned)ofs;
                asm volatile("global_load_async_to_lds_b128 %0, %1, off"
                             :: "v"(la), "v"(ga) : "memory");
            }
        }
    };

    issue(0, 0);
#pragma unroll 1
    for (int nt = 0; nt < NT; ++nt) {
        const int cur = nt & 1;
        if (nt + 1 < NT) {
            issue(nt + 1, cur ^ 1);
            asm volatile("s_wait_asynccnt %0" :: "i"(CPW) : "memory");
        } else {
            asm volatile("s_wait_asynccnt 0" ::: "memory");
        }
        WG_BARRIER();      // chunk nt resident in LDS for all waves

        v8f acc = {};
        const _Float16* wl = &bsh[cur][0] + lane * 16;
#pragma unroll
        for (int kc = 0; kc < KCH; ++kc) {
            v16h bfrag = *(const v16h*)(wl + kc * 512);
            acc = __builtin_amdgcn_wmma_f32_16x16x32_f16(
                false, a[kc].h, false, bfrag, (short)0, acc, false, false);
        }

        // epilogue: bias + ReLU + BN, transpose tile through LDS, b128 stores
        const int col0 = nt * 16;
        const int colL = lane & 15;
        const float bb = bias[col0 + colL];
        const float gg = gamma[col0 + colL] * BN_RS;
        const float be = beta[col0 + colL];
        _Float16* tw = &tsh[wave][0];
        const int rb = (lane < 16) ? 0 : 8;
#pragma unroll
        for (int r = 0; r < 8; ++r) {
            float z = fmaxf(acc[r] + bb, 0.0f);
            z = gg * z + be;
            tw[(rb + r) * 16 + colL] = (_Float16)z;   // wave-private, in-order
        }
        const int rr  = lane >> 1;
        const int seg = (lane & 1) * 8;
        v4u vv = *(const v4u*)(tw + rr * 16 + seg);
        *(v4u*)(Y + (row0 + rr) * (size_t)COUT + col0 + seg) = vv;

        WG_BARRIER();      // all reads of bsh[cur] done before it is reloaded
    }
}

// Pre-pack f32 weights [CIN,COUT] into f16 wave32 B-fragment order
// [(nt*KCH+kc)*512 + lane*16 + h], zero-padding K to CINP.
__global__ void prepack_kernel(const float* __restrict__ W,
                               _Float16* __restrict__ Wpk,
                               int CIN, int COUT, int KCH, int NT)
{
    int i = blockIdx.x * 256 + threadIdx.x;
    if (i >= KCH * NT * 512) return;
    int h = i & 15;
    int lane = (i >> 4) & 31;
    int e = i >> 9;
    int kc = e % KCH;
    int nt = e / KCH;
    int v = h >> 1, p = h & 1;
    int kl = ((lane < 16) ? 0 : 8) + (h & 7) + ((h >> 3) << 4);
    int kk = kc * 32 + kl;
    int n = nt * 16 + (lane & 15);
    float val = (kk < CIN) ? W[(size_t)kk * COUT + n] : 0.0f;
    Wpk[i] = (_Float16)val;
}

// ---------------------------- Pooling --------------------------------------
__global__ void pool_max_kernel(const _Float16* __restrict__ Yk, // [BM,64,C]
                                const int* __restrict__ cnt,
                                _Float16* __restrict__ out, int BM, int C)
{
    size_t i = (size_t)blockIdx.x * 256 + threadIdx.x;
    if (i >= (size_t)BM * C) return;
    int c = (int)(i % C);
    size_t bm = i / C;
    int n = cnt[bm];
    float mx = NEG_INF;
    for (int j = 0; j < n; ++j)
        mx = fmaxf(mx, (float)Yk[((size_t)bm * 64 + j) * C + c]);
    out[i] = (_Float16)mx;
}

__global__ void pool_global_kernel(const _Float16* __restrict__ Yg, // [8,256,1024]
                                   float* __restrict__ headx, int br)
{
    int i = blockIdx.x * 256 + threadIdx.x;
    if (i >= 8 * 1024) return;
    int c = i % 1024;
    int b = i / 1024;
    float mx = NEG_INF;
    for (int p = 0; p < 256; ++p)
        mx = fmaxf(mx, (float)Yg[((size_t)b * 256 + p) * 1024 + c]);
    headx[(size_t)b * 2048 + br * 1024 + c] = mx;
}

// ---------------------------- Head -----------------------------------------
__global__ void linear_kernel(const float* __restrict__ X,
                              const float* __restrict__ W,
                              const float* __restrict__ bias,
                              float* __restrict__ Y, int K, int Cout, int act)
{
    int i = blockIdx.x * 256 + threadIdx.x;
    if (i >= 8 * Cout) return;
    int c = i % Cout;
    int r = i / Cout;
    float acc = bias[c];
    for (int k = 0; k < K; ++k)
        acc += X[(size_t)r * K + k] * W[(size_t)k * Cout + c];
    if (act) acc = fmaxf(acc, 0.0f);
    Y[i] = acc;
}

__global__ void log_softmax_kernel(const float* __restrict__ X,
                                   float* __restrict__ out)
{
    int r = threadIdx.x;
    if (r >= 8) return;
    float mx = NEG_INF;
    for (int c = 0; c < 8; ++c) mx = fmaxf(mx, X[r * 8 + c]);
    float s = 0.0f;
    for (int c = 0; c < 8; ++c) s += __expf(X[r * 8 + c] - mx);
    float l = __logf(s) + mx;
    for (int c = 0; c < 8; ++c) out[r * 8 + c] = X[r * 8 + c] - l;
}

// ---------------------------------------------------------------------------
extern "C" void kernel_launch(void* const* d_in, const int* in_sizes, int n_in,
                              void* d_out, int out_size, void* d_ws, size_t ws_size,
                              hipStream_t stream)
{
    (void)in_sizes; (void)n_in; (void)out_size; (void)ws_size;

    // -------- input pytree: pos1, pos2, then params sorted by key ----------
    const float* posIn[2] = { (const float*)d_in[0], (const float*)d_in[1] };
    const float* linW[4]; const float* linB[4];
    for (int l = 0; l < 4; ++l) {
        linW[l] = (const float*)d_in[2 + 2 * l];
        linB[l] = (const float*)d_in[3 + 2 * l];
    }
    const float *saW[3][3], *saB[3][3], *saG[3][3], *saBE[3][3];
    for (int s = 0; s < 3; ++s)
        for (int l = 0; l < 3; ++l) {
            int base = 10 + s * 12 + l * 4;
            saW[s][l]  = (const float*)d_in[base + 0];
            saB[s][l]  = (const float*)d_in[base + 1];
            saG[s][l]  = (const float*)d_in[base + 2];
            saBE[s][l] = (const float*)d_in[base + 3];
        }

    // ----------------------- workspace layout ------------------------------
    char* ws = (char*)d_ws;
    size_t off = 0;
    auto alloc = [&](size_t bytes) -> char* {
        char* p = ws + off;
        off += (bytes + 255) & ~(size_t)255;
        return p;
    };
    _Float16* buf0  = (_Float16*)alloc((size_t)524288 * 128 * 2); // ping
    _Float16* buf1  = (_Float16*)alloc((size_t)524288 * 64 * 2);  // pong/feat
    int*   idx1  = (int*)  alloc((size_t)8 * 1024 * 4);
    float* c1    = (float*)alloc((size_t)8 * 1024 * 3 * 4);
    int*   nbr1  = (int*)  alloc((size_t)8 * 1024 * 64 * 4);
    int*   cnt1  = (int*)  alloc((size_t)8 * 1024 * 4);
    _Float16* x1 = (_Float16*)alloc((size_t)8 * 1024 * 128 * 2);
    int*   idx2  = (int*)  alloc((size_t)8 * 256 * 4);
    float* c2    = (float*)alloc((size_t)8 * 256 * 3 * 4);
    int*   nbr2  = (int*)  alloc((size_t)8 * 256 * 64 * 4);
    int*   cnt2  = (int*)  alloc((size_t)8 * 256 * 4);
    _Float16* x2 = (_Float16*)alloc((size_t)8 * 256 * 256 * 2);
    float* headx = (float*)alloc((size_t)8 * 2048 * 4);
    float* h1    = (float*)alloc((size_t)8 * 1024 * 4);
    float* h2    = (float*)alloc((size_t)8 * 512 * 4);
    float* h3    = (float*)alloc((size_t)8 * 256 * 4);
    float* h4    = (float*)alloc((size_t)8 * 8 * 4);

    // packed weights: {CIN, CINP, COUT} per layer
    const int cfg[9][3] = {
        {3, 32, 64}, {64, 64, 64}, {64, 64, 128},          // sa1
        {131, 160, 128}, {128, 128, 128}, {128, 128, 256}, // sa2
        {259, 288, 256}, {256, 256, 512}, {512, 512, 1024} // sa3
    };
    _Float16* wpk[9];
    for (int l = 0; l < 9; ++l) {
        int KCH = cfg[l][1] / 32, NT = cfg[l][2] / 16;
        wpk[l] = (_Float16*)alloc((size_t)KCH * NT * 512 * 2);
    }
    for (int l = 0; l < 9; ++l) {
        int KCH = cfg[l][1] / 32, NT = cfg[l][2] / 16;
        int total = KCH * NT * 512;
        prepack_kernel<<<(total + 255) / 256, 256, 0, stream>>>(
            saW[l / 3][l % 3], wpk[l], cfg[l][0], cfg[l][2], KCH, NT);
    }

    // ----------------------- per-branch pipeline ---------------------------
    for (int br = 0; br < 2; ++br) {
        const float* P = posIn[br];

        // SA1: 2048 -> 1024 centers, k=64, r=0.2
        fps_kernel<2048><<<8, 1024, 0, stream>>>(P, 1024, idx1);
        gather_centers_kernel<<<(8 * 1024 * 3 + 255) / 256, 256, 0, stream>>>(
            P, idx1, c1, 2048, 1024);
        group_kernel<2048><<<8 * 1024, 256, 0, stream>>>(
            P, c1, 1024, 0.04f, nbr1, cnt1);
        {
            size_t total = (size_t)8 * 1024 * 64 * 32;
            feat_sa1_kernel<<<(int)((total + 255) / 256), 256, 0, stream>>>(
                P, c1, nbr1, cnt1, buf1);
        }
        mlp_gemm_kernel<32, 64><<<4096, 256, 0, stream>>>(
            buf1, wpk[0], saB[0][0], saG[0][0], saBE[0][0], buf0);
        mlp_gemm_kernel<64, 64><<<4096, 256, 0, stream>>>(
            buf0, wpk[1], saB[0][1], saG[0][1], saBE[0][1], buf1);
        mlp_gemm_kernel<64, 128><<<4096, 256, 0, stream>>>(
            buf1, wpk[2], saB[0][2], saG[0][2], saBE[0][2], buf0);
        pool_max_kernel<<<(8 * 1024 * 128 + 255) / 256, 256, 0, stream>>>(
            buf0, cnt1, x1, 8 * 1024, 128);

        // SA2: 1024 -> 256 centers, k=64, r=0.4
        fps_kernel<1024><<<8, 1024, 0, stream>>>(c1, 256, idx2);
        gather_centers_kernel<<<(8 * 256 * 3 + 255) / 256, 256, 0, stream>>>(
            c1, idx2, c2, 1024, 256);
        group_kernel<1024><<<8 * 256, 256, 0, stream>>>(
            c1, c2, 256, 0.16f, nbr2, cnt2);
        {
            size_t total = (size_t)8 * 256 * 64 * 160;
            feat_sa2_kernel<<<(int)((total + 255) / 256), 256, 0, stream>>>(
                x1, c1, c2, nbr2, cnt2, buf1);
        }
        mlp_gemm_kernel<160, 128><<<1024, 256, 0, stream>>>(
            buf1, wpk[3], saB[1][0], saG[1][0], saBE[1][0], buf0);
        mlp_gemm_kernel<128, 128><<<1024, 256, 0, stream>>>(
            buf0, wpk[4], saB[1][1], saG[1][1], saBE[1][1], buf1);
        mlp_gemm_kernel<128, 256><<<1024, 256, 0, stream>>>(
            buf1, wpk[5], saB[1][2], saG[1][2], saBE[1][2], buf0);
        pool_max_kernel<<<(8 * 256 * 256 + 255) / 256, 256, 0, stream>>>(
            buf0, cnt2, x2, 8 * 256, 256);

        // SA3: global MLP + max over 256 points
        {
            size_t total = (size_t)2048 * 288;
            feat_sa3_kernel<<<(int)((total + 255) / 256), 256, 0, stream>>>(
                x2, c2, buf1);
        }
        mlp_gemm_kernel<288, 256><<<16, 256, 0, stream>>>(
            buf1, wpk[6], saB[2][0], saG[2][0], saBE[2][0], buf0);
        mlp_gemm_kernel<256, 512><<<16, 256, 0, stream>>>(
            buf0, wpk[7], saB[2][1], saG[2][1], saBE[2][1], buf1);
        mlp_gemm_kernel<512, 1024><<<16, 256, 0, stream>>>(
            buf1, wpk[8], saB[2][2], saG[2][2], saBE[2][2], buf0);
        pool_global_kernel<<<(8 * 1024 + 255) / 256, 256, 0, stream>>>(
            buf0, headx, br);
    }

    // ----------------------------- head ------------------------------------
    linear_kernel<<<(8 * 1024 + 255) / 256, 256, 0, stream>>>(
        headx, linW[0], linB[0], h1, 2048, 1024, 1);
    linear_kernel<<<(8 * 512 + 255) / 256, 256, 0, stream>>>(
        h1, linW[1], linB[1], h2, 1024, 512, 1);
    linear_kernel<<<(8 * 256 + 255) / 256, 256, 0, stream>>>(
        h2, linW[2], linB[2], h3, 512, 256, 0);
    linear_kernel<<<1, 64, 0, stream>>>(h3, linW[3], linB[3], h4, 256, 8, 0);
    log_softmax_kernel<<<1, 8, 0, stream>>>(h4, (float*)d_out);
}
